// Head_20066087207129
// MI455X (gfx1250) — matrix operand order repaired and checked by reading, not compiled, not run on hardware
//
#include <hip/hip_runtime.h>

typedef __attribute__((ext_vector_type(16))) _Float16 v16h;
typedef __attribute__((ext_vector_type(8)))  float    v8f;
typedef __attribute__((ext_vector_type(2)))  _Float16 h2;
typedef __attribute__((ext_vector_type(2)))  float    f2;

#define B_  64
#define T_  512
#define C_  384
#define H_  64

// ---------------------------------------------------------------------------
// Kernel 0: W [C,H] f32  ->  WT [3][H][C] f16 (transposed; q-scale folded)
// ---------------------------------------------------------------------------
__global__ __launch_bounds__(256) void wt_prep(const float* __restrict__ Wq,
                                               const float* __restrict__ Wk,
                                               const float* __restrict__ Wv,
                                               _Float16* __restrict__ WT) {
  int idx = blockIdx.x * 256 + threadIdx.x;      // [0, 3*H*C)
  int m = idx / (H_ * C_);
  int r = idx - m * (H_ * C_);
  int h = r / C_;
  int c = r - h * C_;
  const float* W = (m == 0) ? Wq : (m == 1) ? Wk : Wv;
  float v = W[c * H_ + h];
  if (m == 0) v *= 0.125f;                       // 1/sqrt(H), H = 64
  WT[idx] = (_Float16)v;
}

// B fragment 32x16: n = 16*i + lm, k = 2v + 16*hf  (WT is [H][C] row-major)
__device__ __forceinline__ v16h load_bfrag(const _Float16* __restrict__ WT,
                                           int i, int ks, int lm, int hf) {
  v16h bfr;
  const _Float16* Wb = WT + (i * 16 + lm) * C_ + 32 * ks + 16 * hf;
#pragma unroll
  for (int v = 0; v < 8; ++v) {
    h2 wv = *(const h2*)(Wb + 2 * v);
    bfr[2 * v] = wv[0]; bfr[2 * v + 1] = wv[1];
  }
  return bfr;
}

// ---------------------------------------------------------------------------
// Kernel 1: QKV projection.  One wave per 16-row tile of [B*T].
// Rotating double-buffer on B fragments (load b_next before the WMMA that
// consumes b_cur) and deferred-convert double-buffer on the A tile, so each
// WMMA waits with the following fragment already in flight.
// ---------------------------------------------------------------------------
__global__ __launch_bounds__(256) void qkv_proj(const float* __restrict__ x,
                                                const _Float16* __restrict__ WT,
                                                _Float16* __restrict__ qh,
                                                _Float16* __restrict__ kh,
                                                _Float16* __restrict__ vT) {
  const int lane = threadIdx.x & 31;
  const int wave = threadIdx.x >> 5;
  const int tile = blockIdx.x * 8 + wave;        // 16-row tile index
  const int lm   = lane & 15;
  const int hf   = lane >> 4;
  const int arow = tile * 16 + lm;               // row gathered for A-frags

  v8f acc[12] = {};                              // [3 matrices][4 n-tiles]

  // Preload A(ks=0) and convert; preload B(id=0).
  f2 araw[8];
#pragma unroll
  for (int v = 0; v < 8; ++v) {
    int k = 2 * v + 16 * hf;
    araw[v] = *(const f2*)(x + arow * C_ + k);
  }
  v16h a_cur;
#pragma unroll
  for (int v = 0; v < 8; ++v) {
    a_cur[2 * v] = (_Float16)araw[v][0]; a_cur[2 * v + 1] = (_Float16)araw[v][1];
  }
  v16h b_cur = load_bfrag(WT, 0, 0, lm, hf);

#pragma unroll 1
  for (int ks = 0; ks < 12; ++ks) {              // K = 384 = 12 * 32
    // Issue next A-tile x-loads now; convert only after the WMMA chain.
    if (ks < 11) {
#pragma unroll
      for (int v = 0; v < 8; ++v) {
        int k = 32 * (ks + 1) + 2 * v + 16 * hf;
        araw[v] = *(const f2*)(x + arow * C_ + k);
      }
    }
#pragma unroll
    for (int i = 0; i < 12; ++i) {
      int ni  = (i + 1) % 12;                    // folds after unroll
      int nks = (i == 11) ? (ks < 11 ? ks + 1 : 11) : ks;
      v16h b_next = load_bfrag(WT, ni, nks, lm, hf);
      acc[i] = __builtin_amdgcn_wmma_f32_16x16x32_f16(
          false, a_cur, false, b_cur, (short)0, acc[i], false, false);
      b_cur = b_next;
    }
    if (ks < 11) {
#pragma unroll
      for (int v = 0; v < 8; ++v) {
        a_cur[2 * v] = (_Float16)araw[v][0]; a_cur[2 * v + 1] = (_Float16)araw[v][1];
      }
    }
  }

  // C-layout scatter: row = v + 8*hf, col = 16t + lm
#pragma unroll
  for (int t = 0; t < 4; ++t) {
#pragma unroll
    for (int v = 0; v < 8; ++v) {
      int gr = tile * 16 + v + 8 * hf;
      int h  = 16 * t + lm;
      qh[gr * H_ + h] = (_Float16)acc[t][v];
      kh[gr * H_ + h] = (_Float16)acc[4 + t][v];
      int b = gr >> 9, tr = gr & 511;
      vT[(b * H_ + h) * T_ + tr] = (_Float16)acc[8 + t][v];
    }
  }
}

// ---------------------------------------------------------------------------
// Kernel 2: causal flash-attention.  One wave per 16-query tile, 32 keys/iter.
// K and V fragments are both loaded at the top of the loop: V-load latency is
// hidden behind the S-WMMAs and the softmax VALU work.
// ---------------------------------------------------------------------------
__global__ __launch_bounds__(128) void attn(const _Float16* __restrict__ qh,
                                            const _Float16* __restrict__ kh,
                                            const _Float16* __restrict__ vT,
                                            float* __restrict__ out) {
  __shared__ _Float16 psm[4][16][32];            // per-wave P transpose buffer
  const int lane = threadIdx.x & 31;
  const int wave = threadIdx.x >> 5;
  const int lm = lane & 15, hf = lane >> 4;
  const int qtile = blockIdx.x * 4 + wave;
  const int r0g = qtile * 16;                    // global row base
  const int b   = r0g >> 9;
  const int r0  = r0g & 511;                     // row base within sequence

  // Q fragments (scaled at projection time), contraction h split 2 x 32
  v16h qf[2];
#pragma unroll
  for (int c = 0; c < 2; ++c)
#pragma unroll
    for (int v = 0; v < 8; ++v) {
      int h = 32 * c + 2 * v + 16 * hf;
      h2 t = *(const h2*)(qh + (r0g + lm) * H_ + h);
      qf[c][2 * v] = t[0]; qf[c][2 * v + 1] = t[1];
    }

  v8f acc[4] = {};                               // O tile 16x64, 4 n-tiles
  float mrow[8], lrow[8];
#pragma unroll
  for (int v = 0; v < 8; ++v) { mrow[v] = -3.0e38f; lrow[v] = 0.0f; }

  const _Float16* kb = kh + b * (T_ * H_);
  const _Float16* vb = vT + b * (H_ * T_);

  for (int j0 = 0; j0 <= r0 + 15; j0 += 32) {
    {                                            // prefetch next K/V tiles
      int jn = j0 + 32; if (jn > T_ - 32) jn = T_ - 32;
      __builtin_prefetch(kb + jn * H_ + lane * 64, 0, 1);
      __builtin_prefetch(vb + lane * T_ + jn, 0, 1);
      __builtin_prefetch(vb + (32 + lane) * T_ + jn, 0, 1);
    }

    // ---- issue ALL fragment loads first (K then V) ----
    v16h kf[2][2];                               // [subtile][h-chunk]
#pragma unroll
    for (int st = 0; st < 2; ++st) {
      const _Float16* kr = kb + (j0 + 16 * st + lm) * H_ + 16 * hf;
#pragma unroll
      for (int v = 0; v < 8; ++v) {
        h2 t0 = *(const h2*)(kr + 2 * v);
        h2 t1 = *(const h2*)(kr + 32 + 2 * v);
        kf[st][0][2 * v] = t0[0]; kf[st][0][2 * v + 1] = t0[1];
        kf[st][1][2 * v] = t1[0]; kf[st][1][2 * v + 1] = t1[1];
      }
    }
    v16h vf[4];                                  // V B-frags: n = 16t+lm, k = key
#pragma unroll
    for (int t = 0; t < 4; ++t) {
      const _Float16* vr = vb + (t * 16 + lm) * T_ + j0 + 16 * hf;
#pragma unroll
      for (int v = 0; v < 8; ++v) {
        h2 tv = *(const h2*)(vr + 2 * v);
        vf[t][2 * v] = tv[0]; vf[t][2 * v + 1] = tv[1];
      }
    }

    // ---- S = Q * K^T (contraction h = 64 = 2 chained WMMAs per subtile) ----
    v8f s[2];
#pragma unroll
    for (int st = 0; st < 2; ++st) {
      v8f c = {};
      c = __builtin_amdgcn_wmma_f32_16x16x32_f16(false, qf[0], false, kf[st][0],
                                                 (short)0, c, false, false);
      c = __builtin_amdgcn_wmma_f32_16x16x32_f16(false, qf[1], false, kf[st][1],
                                                 (short)0, c, false, false);
      s[st] = c;
    }

    // ---- online softmax (row = v + 8*hf spans the 16 lanes of a half) ----
#pragma unroll
    for (int v = 0; v < 8; ++v) {
      int i = r0 + v + 8 * hf;
      float s0 = (j0 + lm      <= i) ? s[0][v] : -3.0e38f;
      float s1 = (j0 + 16 + lm <= i) ? s[1][v] : -3.0e38f;
      float tm = fmaxf(s0, s1);
#pragma unroll
      for (int d = 8; d >= 1; d >>= 1) tm = fmaxf(tm, __shfl_xor(tm, d, 16));
      float mn = fmaxf(mrow[v], tm);
      float al = __expf(mrow[v] - mn);
      float p0 = __expf(s0 - mn);
      float p1 = __expf(s1 - mn);
      float rs = p0 + p1;
#pragma unroll
      for (int d = 8; d >= 1; d >>= 1) rs += __shfl_xor(rs, d, 16);
      lrow[v] = lrow[v] * al + rs;
      mrow[v] = mn;
#pragma unroll
      for (int t = 0; t < 4; ++t) acc[t][v] *= al;
      psm[wave][v + 8 * hf][lm]      = (_Float16)p0;   // C-layout -> LDS
      psm[wave][v + 8 * hf][lm + 16] = (_Float16)p1;
    }

    // Same-wave DS ops are in-order; wait, then reload P in A-layout
    asm volatile("s_wait_dscnt 0x0" ::: "memory");
    v16h pf;
#pragma unroll
    for (int v = 0; v < 8; ++v) {
      int kk = 2 * v + 16 * hf;
      h2 t = *(const h2*)(&psm[wave][lm][kk]);
      pf[2 * v] = t[0]; pf[2 * v + 1] = t[1];
    }

    // ---- O += P * V (V frags already resident) ----
#pragma unroll
    for (int t = 0; t < 4; ++t)
      acc[t] = __builtin_amdgcn_wmma_f32_16x16x32_f16(false, pf, false, vf[t],
                                                      (short)0, acc[t], false, false);
  }

  // Normalize and write out (f32, [B*T][H])
#pragma unroll
  for (int t = 0; t < 4; ++t)
#pragma unroll
    for (int v = 0; v < 8; ++v)
      out[(r0g + v + 8 * hf) * H_ + 16 * t + lm] = acc[t][v] / lrow[v];
}

// ---------------------------------------------------------------------------
extern "C" void kernel_launch(void* const* d_in, const int* in_sizes, int n_in,
                              void* d_out, int out_size, void* d_ws, size_t ws_size,
                              hipStream_t stream) {
  const float* x  = (const float*)d_in[0];
  const float* Wq = (const float*)d_in[1];
  const float* Wk = (const float*)d_in[2];
  const float* Wv = (const float*)d_in[3];

  char* ws = (char*)d_ws;
  const size_t WT_BYTES = (size_t)3 * H_ * C_ * 2;          // 147456
  const size_t QKV_BYTES = (size_t)B_ * T_ * H_ * 2;        // 4 MiB each
  _Float16* WT = (_Float16*)ws;
  _Float16* qh = (_Float16*)(ws + WT_BYTES);
  _Float16* kh = (_Float16*)(ws + WT_BYTES + QKV_BYTES);
  _Float16* vT = (_Float16*)(ws + WT_BYTES + 2 * QKV_BYTES);
  float* out = (float*)d_out;

  wt_prep<<<(3 * H_ * C_) / 256, 256, 0, stream>>>(Wq, Wk, Wv, WT);
  qkv_proj<<<(B_ * T_ / 16) / 8, 256, 0, stream>>>(x, WT, qh, kh, vT);
  attn<<<(B_ * T_ / 16) / 4, 128, 0, stream>>>(qh, kh, vT, out);
}
